// CTTGNN_72902774882529
// MI455X (gfx1250) — matrix operand order repaired and checked by reading, not compiled, not run on hardware
//
#include <hip/hip_runtime.h>
#include <math.h>

// ---------------------------------------------------------------------------
// CTTGNN on gfx1250: register-direct f16 WMMA GEMMs (f32 accum) + L2-resident
// edge scatter. Exact linear-algebra rewrites:
//   h[src]@W_msg               -> (h@W_msg)[src]             (gather of 10MB, L2)
//   segment_sum(ef@W_edge,dst) -> segment_sum(ef,dst)@W_edge (precomputed base)
// Weights are pre-transposed/padded to [Npad][Kpad] f16 so both WMMA fragments
// load as contiguous b128 global loads. Each wave computes 4 M-tiles x 1 N-tile,
// reusing the B fragment across 4 independent WMMA chains.
// ---------------------------------------------------------------------------

typedef __attribute__((ext_vector_type(16))) _Float16 v16h;
typedef __attribute__((ext_vector_type(8)))  _Float16 v8h;
typedef __attribute__((ext_vector_type(8)))  float    v8f;

#define WPB 8   // waves per block (256 threads, wave32)
#define MPW 4   // 16-row M tiles per wave

// C[M,N] = act(A[M,Kpad](f16,row-major,16B-aligned rows) @ B + bias)
// BT = B transposed+padded: [Npad>=16][Kpad], row n = column n of B.
// KT = Kpad (compile-time, multiple of 32). act: 0 none,1 ELU,2 ReLU,3 sigmoid.
template <int KT>
__global__ __launch_bounds__(256) void k_gemm_t(
    const _Float16* __restrict__ A, int lda,
    const _Float16* __restrict__ BT, int ldbt,
    const float* __restrict__ bias,
    float* __restrict__ Cf, _Float16* __restrict__ Ch, int ldc,
    int M, int N, int act)
{
  const int lane = threadIdx.x & 31;
  const int tilesN  = (N + 15) >> 4;
  const int tilesM4 = (M + (MPW * 16 - 1)) / (MPW * 16);
  const int nTiles  = tilesM4 * tilesN;
  int tileId = blockIdx.x * WPB + (threadIdx.x >> 5);
  const bool active = (tileId < nTiles);
  if (!active) tileId = 0;
  const int tM = (tileId / tilesN) * (MPW * 16);
  const int tN = (tileId % tilesN) << 4;

  // A fragments: lane covers row (tMi + lane&15); two contiguous 8-half runs
  // per 32-K chunk at K offsets akb and akb+16.
  const int am  = lane & 15;
  const int akb = (lane < 16) ? 0 : 8;
  const _Float16* Arow[MPW];
#pragma unroll
  for (int mi = 0; mi < MPW; ++mi) {
    int ar = tM + mi * 16 + am;
    if (ar >= M) ar = M - 1;  // clamp -> loads in bounds; stores are masked
    Arow[mi] = A + (size_t)ar * lda + akb;
  }

  // B fragment: lane covers column (tN + lane&15) of B, i.e. one row of BT;
  // two contiguous 8-half runs per 32-K chunk at bkb and bkb+16. Reused by all MPW.
  const int bn  = lane & 15;
  const int bkb = (lane < 16) ? 0 : 8;
  const _Float16* Brow = BT + (size_t)(tN + bn) * ldbt + bkb;

  v8f c[MPW];
#pragma unroll
  for (int mi = 0; mi < MPW; ++mi) c[mi] = (v8f){};

#pragma unroll
  for (int k0 = 0; k0 < KT; k0 += 32) {
    const v8h b0 = *(const v8h*)(Brow + k0);
    const v8h b1 = *(const v8h*)(Brow + k0 + 16);
    const v16h b = __builtin_shufflevector(b0, b1, 0, 1, 2, 3, 4, 5, 6, 7,
                                           8, 9, 10, 11, 12, 13, 14, 15);
#pragma unroll
    for (int mi = 0; mi < MPW; ++mi) {
      const v8h a0 = *(const v8h*)(Arow[mi] + k0);
      const v8h a1 = *(const v8h*)(Arow[mi] + k0 + 16);
      const v16h a = __builtin_shufflevector(a0, a1, 0, 1, 2, 3, 4, 5, 6, 7,
                                             8, 9, 10, 11, 12, 13, 14, 15);
      c[mi] = __builtin_amdgcn_wmma_f32_16x16x32_f16(false, a, false, b,
                                                     (short)0, c[mi],
                                                     false, false);
    }
  }

  if (active) {
    const int n = tN + (lane & 15);
    if (n < N) {
      const float bv = bias ? bias[n] : 0.f;
#pragma unroll
      for (int mi = 0; mi < MPW; ++mi) {
#pragma unroll
        for (int vv = 0; vv < 8; ++vv) {
          const int m = tM + mi * 16 + vv + ((lane < 16) ? 0 : 8);
          if (m < M) {
            float val = c[mi][vv] + bv;
            if (act == 1)      val = val > 0.f ? val : (__expf(val) - 1.f);
            else if (act == 2) val = val > 0.f ? val : 0.f;
            else if (act == 3) val = 1.f / (1.f + __expf(-val));
            if (Cf) Cf[(size_t)m * ldc + n] = val;
            if (Ch) Ch[(size_t)m * ldc + n] = (_Float16)val;
          }
        }
      }
    }
  }
}

// LayerNorm over width D (<=128), one wave per row.
__global__ __launch_bounds__(256) void k_ln(
    const float* __restrict__ X, int ldx,
    float* __restrict__ Yf, _Float16* __restrict__ Yh, int ldy,
    const float* __restrict__ g, const float* __restrict__ bta,
    int M, int D)
{
  const int row  = blockIdx.x * WPB + (threadIdx.x >> 5);
  const int lane = threadIdx.x & 31;
  if (row >= M) return;
  const float* x = X + (size_t)row * ldx;
  const int per = (D + 31) >> 5;  // 1 or 4
  float vals[4];
  float s = 0.f;
  for (int j = 0; j < per; ++j) {
    const int idx = lane + (j << 5);
    vals[j] = (idx < D) ? x[idx] : 0.f;
    s += vals[j];
  }
  for (int m = 16; m; m >>= 1) s += __shfl_xor(s, m, 32);
  const float mean = s / (float)D;
  float v2 = 0.f;
  for (int j = 0; j < per; ++j) {
    const int idx = lane + (j << 5);
    const float d = (idx < D) ? (vals[j] - mean) : 0.f;
    v2 += d * d;
  }
  for (int m = 16; m; m >>= 1) v2 += __shfl_xor(v2, m, 32);
  const float inv = rsqrtf(v2 / (float)D + 1e-5f);
  for (int j = 0; j < per; ++j) {
    const int idx = lane + (j << 5);
    if (idx < D) {
      const float o = (vals[j] - mean) * inv * g[idx] + bta[idx];
      if (Yf) Yf[(size_t)row * ldy + idx] = o;
      if (Yh) Yh[(size_t)row * ldy + idx] = (_Float16)o;
    }
  }
}

// out[sidx[e]] += src[gidx?gidx[e]:e]  (128 floats/row, 4 threads/edge)
__global__ __launch_bounds__(256) void k_scatter_add(
    const float* __restrict__ src, const int* __restrict__ gidx,
    const int* __restrict__ sidx, float* __restrict__ out, int E)
{
  const int t = blockIdx.x * blockDim.x + threadIdx.x;
  const int e = t >> 2;
  if (e >= E) return;
  const int part = (t & 3) << 5;
  if (((e & 63) == 0) && (e + 512 < E)) {
    __builtin_prefetch(&sidx[e + 512], 0, 1);
    if (gidx) __builtin_prefetch(&gidx[e + 512], 0, 1);
  }
  const int gs = gidx ? gidx[e] : e;
  const int ds = sidx[e];
  const float4* s4 = (const float4*)(src + (size_t)gs * 128 + part);
  float* o = out + (size_t)ds * 128 + part;
#pragma unroll
  for (int j = 0; j < 8; ++j) {
    const float4 v = s4[j];
    atomicAdd(o + 4 * j + 0, v.x);
    atomicAdd(o + 4 * j + 1, v.y);
    atomicAdd(o + 4 * j + 2, v.z);
    atomicAdd(o + 4 * j + 3, v.w);
  }
}

// out = ca*a + cb*(dtflag ? (ts[t0i+1]-ts[t0i]) : 1)*b
__global__ void k_combine(float* __restrict__ out, _Float16* __restrict__ outh,
                          const float* __restrict__ a, const float* __restrict__ b,
                          float ca, float cb, const float* __restrict__ ts,
                          int t0i, int dtflag, int n)
{
  const int i = blockIdx.x * blockDim.x + threadIdx.x;
  if (i >= n) return;
  float coef = cb;
  if (dtflag) coef *= (ts[t0i + 1] - ts[t0i]);
  const float v = ca * a[i] + coef * b[i];
  out[i] = v;
  if (outh) outh[i] = (_Float16)v;
}

__global__ void k_tanh2(float* __restrict__ out, _Float16* __restrict__ outh,
                        const float* __restrict__ hS, const float* __restrict__ agg, int n)
{
  const int i = blockIdx.x * blockDim.x + threadIdx.x;
  if (i >= n) return;
  const float v = tanhf(hS[i] + agg[i]);
  out[i] = v;
  if (outh) outh[i] = (_Float16)v;
}

// multi-scale gate: softmax over 4 scale logits, sin(t/tau) gates, mix feats.
__global__ void k_msgate(float* __restrict__ hf, _Float16* __restrict__ hh,
                         const float* __restrict__ f0, const float* __restrict__ f1,
                         const float* __restrict__ f2, const float* __restrict__ f3,
                         const float* __restrict__ logits,
                         const float* __restrict__ ts, int tidx, int n)
{
  const int i = blockIdx.x * blockDim.x + threadIdx.x;
  if (i >= n) return;
  const int r = i >> 7;  // H = 128
  const float t1 = ts[tidx];
  const float l0 = logits[r * 4 + 0], l1 = logits[r * 4 + 1];
  const float l2 = logits[r * 4 + 2], l3 = logits[r * 4 + 3];
  const float mx = fmaxf(fmaxf(l0, l1), fmaxf(l2, l3));
  const float e0 = __expf(l0 - mx), e1 = __expf(l1 - mx);
  const float e2 = __expf(l2 - mx), e3 = __expf(l3 - mx);
  const float inv = 1.f / (e0 + e1 + e2 + e3);
  const float s0 = __sinf(t1 * 1e6f), s1 = __sinf(t1 * 1e3f);
  const float s2 = __sinf(t1),        s3 = __sinf(t1 / 3600.f);
  const float v = (f0[i] * s0 * e0 + f1[i] * s1 * e1 +
                   f2[i] * s2 * e2 + f3[i] * s3 * e3) * inv;
  hf[i] = v;
  hh[i] = (_Float16)v;
}

__global__ void k_cvt(const float* __restrict__ in, _Float16* __restrict__ out, int n)
{
  const int i = blockIdx.x * blockDim.x + threadIdx.x;
  if (i < n) out[i] = (_Float16)in[i];
}

// Transposed + padded weight conversion: in = B[K][N] f32 (row-major),
// out = BT[Npad][Kpad] f16, out[n][k] = (k<K && n<N) ? B[k][n] : 0
__global__ void k_cvt_t(const float* __restrict__ in, _Float16* __restrict__ out,
                        int K, int N, int Kpad, int total)
{
  const int i = blockIdx.x * blockDim.x + threadIdx.x;
  if (i >= total) return;
  const int n = i / Kpad;
  const int k = i - n * Kpad;
  const float v = (k < K && n < N) ? in[(size_t)k * N + n] : 0.f;
  out[i] = (_Float16)v;
}

// Repack edge_attr [E][87] f32 into aligned padded f16 [E][128]:
// part p (K={20,30,25,12}, src col {0,20,50,75}) at column p*32, zero padded.
__global__ void k_pad_edge(const float* __restrict__ ea, _Float16* __restrict__ out, int total)
{
  const int i = blockIdx.x * blockDim.x + threadIdx.x;
  if (i >= total) return;
  const int e = i >> 7, col = i & 127;
  const int p = col >> 5, c = col & 31;
  const int Kp[4]  = {20, 30, 25, 12};
  const int ofs[4] = {0, 20, 50, 75};
  const float v = (c < Kp[p]) ? ea[(size_t)e * 87 + ofs[p] + c] : 0.f;
  out[i] = (_Float16)v;
}

__global__ void k_zero(float* __restrict__ p, int n)
{
  const int i = blockIdx.x * blockDim.x + threadIdx.x;
  if (i < n) p[i] = 0.f;
}

// ---------------------------------------------------------------------------
extern "C" void kernel_launch(void* const* d_in, const int* in_sizes, int n_in,
                              void* d_out, int out_size, void* d_ws, size_t ws_size,
                              hipStream_t stream)
{
  (void)in_sizes; (void)out_size; (void)ws_size;
  constexpr int N = 20000, E = 320000, H = 128;

  const float* x    = (const float*)d_in[0];
  const int*   eidx = (const int*)d_in[1];
  const float* ea   = (const float*)d_in[2];
  const float* ts   = (const float*)d_in[3];
  const int* src = eidx;
  const int* dst = eidx + E;

  // params flattened in jax pytree (sorted-key) order, offset by 4 inputs
  const float* P[70] = {};
  for (int i = 0; i < n_in && i < 70; ++i) P[i] = (const float*)d_in[i];
  // 4..7  cls.l1{b,w} cls.l2{b,w}
  // 8..27 edge_enc dir/flow/fusion/timing/tls: lin.b, lin.w, ln.b, ln.g
  // 28,29 ms.attn{b,w}; 30..45 ms.transforms[0..3]{l1.b,l1.w,l2.b,l2.w}
  // 46..49 node_enc lin.b, lin.w, ln.b, ln.g
  // 50..65 ode[blk].layers[l]: b, w_edge, w_msg, w_self  (blk*2+l)
  // 66..69 threat.l1{b,w} threat.l2{b,w}

  char* wsb = (char*)d_ws;
  size_t off = 0;
  auto alloc = [&](size_t bytes) -> void* {
    void* p = wsb + off;
    off += (bytes + 255) & ~(size_t)255;
    return p;
  };
  auto cvt = [&](const float* in, _Float16* out, int n) {
    k_cvt<<<(n + 255) / 256, 256, 0, stream>>>(in, out, n);
  };

  // --- pre-transposed, padded f16 weights (persist) ---
  struct WInfo { int idx, K, Nn; };
  static const WInfo winfos[] = {
    {5, 128, 64}, {7, 64, 2}, {9, 12, 32}, {13, 25, 32}, {17, 128, 128},
    {21, 20, 32}, {25, 30, 32}, {29, 128, 4},
    {31, 128, 128}, {33, 128, 128}, {35, 128, 128}, {37, 128, 128},
    {39, 128, 128}, {41, 128, 128}, {43, 128, 128}, {45, 128, 128},
    {47, 64, 128},
    {51, 128, 128}, {52, 128, 128}, {53, 128, 128},
    {55, 128, 128}, {56, 128, 128}, {57, 128, 128},
    {59, 128, 128}, {60, 128, 128}, {61, 128, 128},
    {63, 128, 128}, {64, 128, 128}, {65, 128, 128},
    {67, 128, 32}, {69, 32, 1}
  };
  _Float16* WT[70] = {};
  int WKpad[70] = {};
  for (const auto& w : winfos) {
    const int Kpad = (w.K + 31) & ~31;
    const int Npad = (w.Nn + 15) & ~15;
    const int tot  = Npad * Kpad;
    WT[w.idx] = (_Float16*)alloc((size_t)tot * sizeof(_Float16));
    WKpad[w.idx] = Kpad;
    k_cvt_t<<<(tot + 255) / 256, 256, 0, stream>>>(P[w.idx], WT[w.idx],
                                                   w.K, w.Nn, Kpad, tot);
  }

  // gemm: dispatch on compile-time Kpad so K-loop fully unrolls.
  auto gemm = [&](const _Float16* A, int lda, int wIdx, const float* bias,
                  float* Cf, _Float16* Ch, int ldc, int M, int NN, int act) {
    const int tiles = ((M + MPW * 16 - 1) / (MPW * 16)) * ((NN + 15) / 16);
    const int blocks = (tiles + WPB - 1) / WPB;
    const _Float16* BT = WT[wIdx];
    const int ldbt = WKpad[wIdx];
    switch (ldbt) {
      case 32:
        k_gemm_t<32><<<blocks, 256, 0, stream>>>(A, lda, BT, ldbt, bias,
                                                 Cf, Ch, ldc, M, NN, act);
        break;
      case 64:
        k_gemm_t<64><<<blocks, 256, 0, stream>>>(A, lda, BT, ldbt, bias,
                                                 Cf, Ch, ldc, M, NN, act);
        break;
      default:
        k_gemm_t<128><<<blocks, 256, 0, stream>>>(A, lda, BT, ldbt, bias,
                                                  Cf, Ch, ldc, M, NN, act);
        break;
    }
  };

  // --- persistent node-sized buffers ---
  float*    S   = (float*)alloc((size_t)N * H * 4);
  _Float16* S_h = (_Float16*)alloc((size_t)N * H * 2);
  float* baseB[4];
  for (int i = 0; i < 4; ++i) baseB[i] = (float*)alloc((size_t)N * H * 4);

  // ===================== Phase 1: edge encoder (big overlay) ================
  const size_t overlay = off;
  _Float16* eap    = (_Float16*)alloc((size_t)E * H * 2);  // padded parts, f16
  float*    pcat   = (float*)alloc((size_t)E * H * 4);
  _Float16* pcat_h = (_Float16*)alloc((size_t)E * H * 2);

  k_pad_edge<<<(E * H + 255) / 256, 256, 0, stream>>>(ea, eap, E * H);
  struct Part { int wI, bI, lnB, lnG, colOut; };
  const Part parts[4] = {
    {21, 20, 22, 23,  0},   // timing (K=20)
    {25, 24, 26, 27, 32},   // tls    (K=30)
    {13, 12, 14, 15, 64},   // flow   (K=25)
    { 9,  8, 10, 11, 96},   // dir    (K=12)
  };
  for (int p = 0; p < 4; ++p) {
    const Part& q = parts[p];
    gemm(eap + p * 32, H, q.wI, P[q.bI], pcat + q.colOut, nullptr, H, E, 32, 1);
    k_ln<<<(E + WPB - 1) / WPB, 256, 0, stream>>>(
        pcat + q.colOut, H, pcat + q.colOut, pcat_h + q.colOut, H,
        P[q.lnG], P[q.lnB], E, 32);
  }
  // fusion: ef = LN(ELU(pcat @ Wf + b))
  gemm(pcat_h, H, 17, P[16], pcat, nullptr, H, E, H, 1);
  k_ln<<<(E + WPB - 1) / WPB, 256, 0, stream>>>(
      pcat, H, pcat, nullptr, H, P[19], P[18], E, H);
  // S = segment_sum(ef, dst)
  k_zero<<<(N * H + 255) / 256, 256, 0, stream>>>(S, N * H);
  k_scatter_add<<<(E * 4 + 255) / 256, 256, 0, stream>>>(pcat, nullptr, dst, S, E);
  cvt(S, S_h, N * H);
  // base[blk*2+l] = S @ w_edge   (exact: segment_sum(ef@W) = segment_sum(ef)@W)
  for (int i = 0; i < 4; ++i)
    gemm(S_h, H, 50 + i * 4 + 1, nullptr, baseB[i], nullptr, H, N, H, 0);

  // ===================== Phase 2: node state overlay ========================
  off = overlay;  // edge-encode buffers dead; reuse region
  _Float16* x_h   = (_Float16*)alloc((size_t)N * 64 * 2);
  float* h        = (float*)alloc((size_t)N * H * 4);
  _Float16* h_h   = (_Float16*)alloc((size_t)N * H * 2);
  float* y        = (float*)alloc((size_t)N * H * 4);
  _Float16* y_h   = (_Float16*)alloc((size_t)N * H * 2);
  float* z        = (float*)alloc((size_t)N * H * 4);
  _Float16* z_h   = (_Float16*)alloc((size_t)N * H * 2);
  float* kb       = (float*)alloc((size_t)N * H * 4);
  float* acc      = (float*)alloc((size_t)N * H * 4);
  float* hW       = (float*)alloc((size_t)N * H * 4);
  float* hS       = (float*)alloc((size_t)N * H * 4);
  float* agg      = (float*)alloc((size_t)N * H * 4);
  float* tmp1     = (float*)alloc((size_t)N * H * 4);
  _Float16* tmp1h = (_Float16*)alloc((size_t)N * H * 2);
  float* feats[4];
  for (int i = 0; i < 4; ++i) feats[i] = (float*)alloc((size_t)N * H * 4);
  float* alpha    = (float*)alloc((size_t)N * 4 * 4);

  // node encoder: h = LN(ELU(x @ W + b))
  cvt(x, x_h, N * 64);
  gemm(x_h, 64, 47, P[46], tmp1, nullptr, H, N, H, 1);
  k_ln<<<(N + WPB - 1) / WPB, 256, 0, stream>>>(
      tmp1, H, h, h_h, H, P[49], P[48], N, H);

  const int NH = N * H;
  auto combine = [&](float* out, _Float16* outh, const float* a, const float* b,
                     float ca, float cb, int t0i, int dtf) {
    k_combine<<<(NH + 255) / 256, 256, 0, stream>>>(out, outh, a, b, ca, cb,
                                                    ts, t0i, dtf, NH);
  };
  auto feval = [&](int blk, const _Float16* in_h, float* outk) {
    const _Float16* cur = in_h;
    for (int l = 0; l < 2; ++l) {
      const int pb = 50 + (blk * 2 + l) * 4;
      gemm(cur, H, pb + 2, nullptr, hW, nullptr, H, N, H, 0);  // h@w_msg
      hipMemcpyAsync(agg, baseB[blk * 2 + l], (size_t)NH * 4,
                     hipMemcpyDeviceToDevice, stream);
      k_scatter_add<<<(E * 4 + 255) / 256, 256, 0, stream>>>(hW, src, dst, agg, E);
      gemm(cur, H, pb + 3, P[pb], hS, nullptr, H, N, H, 0);    // h@w_self + b
      float*    of = (l == 0) ? z : outk;
      _Float16* oh = (l == 0) ? z_h : nullptr;
      k_tanh2<<<(NH + 255) / 256, 256, 0, stream>>>(of, oh, hS, agg, NH);
      cur = z_h;
    }
  };
  auto rk4block = [&](int blk, int t) {
    for (int s = 0; s < 2; ++s) {  // RK4_STEPS = 2; dt = dt_total/2
      feval(blk, h_h, kb);                                   // k1
      combine(acc, nullptr, kb, kb, 1.f, 0.f, t, 0);         // acc = k1
      combine(y, y_h, h, kb, 1.f, 0.25f, t, 1);              // h + 0.5*dt*k1
      feval(blk, y_h, kb);                                   // k2
      combine(acc, nullptr, acc, kb, 1.f, 2.f, t, 0);
      combine(y, y_h, h, kb, 1.f, 0.25f, t, 1);              // h + 0.5*dt*k2
      feval(blk, y_h, kb);                                   // k3
      combine(acc, nullptr, acc, kb, 1.f, 2.f, t, 0);
      combine(y, y_h, h, kb, 1.f, 0.5f, t, 1);               // h + dt*k3
      feval(blk, y_h, kb);                                   // k4
      combine(acc, nullptr, acc, kb, 1.f, 1.f, t, 0);
      combine(h, h_h, h, acc, 1.f, 1.f / 12.f, t, 1);        // h + (dt/6)*acc
    }
  };

  // time loop (T-1 = 2 intervals, host-unrolled; dt read from device ts[])
  for (int t = 0; t < 2; ++t) {
    for (int s = 0; s < 4; ++s) {
      const int mb = 30 + s * 4;
      gemm(h_h, H, mb + 1, P[mb], tmp1, tmp1h, H, N, H, 1);
      gemm(tmp1h, H, mb + 3, P[mb + 2], feats[s], nullptr, H, N, H, 0);
    }
    gemm(h_h, H, 29, P[28], alpha, nullptr, 4, N, 4, 0);
    k_msgate<<<(NH + 255) / 256, 256, 0, stream>>>(
        h, h_h, feats[0], feats[1], feats[2], feats[3], alpha, ts, t + 1, NH);
    rk4block(0, t);
    rk4block(1, t);
  }

  // heads -> d_out = [logits N*2 | threat N | h N*H]
  float* outLogits = (float*)d_out;
  float* outThreat = outLogits + (size_t)N * 2;
  float* outH      = outThreat + (size_t)N;
  gemm(h_h, H, 5, P[4], tmp1, tmp1h, 64, N, 64, 1);            // ELU
  gemm(tmp1h, 64, 7, P[6], outLogits, nullptr, 2, N, 2, 0);
  gemm(h_h, H, 67, P[66], tmp1, tmp1h, 32, N, 32, 2);          // ReLU
  gemm(tmp1h, 32, 69, P[68], outThreat, nullptr, 1, N, 1, 3);  // sigmoid
  hipMemcpyAsync(outH, h, (size_t)NH * 4, hipMemcpyDeviceToDevice, stream);
}
